// BasicBlock_67027259621652
// MI455X (gfx1250) — 8-bit matrix operand order repaired and checked by reading, compiled, not run on hardware
//
#include <hip/hip_runtime.h>

typedef int v8i __attribute__((ext_vector_type(8)));
typedef int vint4_t __attribute__((vector_size(16)));

#define CIN     32
#define COUT    32
#define HW      224
#define NB      16
#define TAPS    10
#define XCOLS   226
#define THREADS 448
#define WAVES   14
#define BN_EPS  1e-5f

#if defined(__has_builtin)
#if __has_builtin(__builtin_amdgcn_global_load_async_to_lds_b128)
#define HAVE_ASYNC_LDS 1
#endif
#endif

__global__ void init_accums(unsigned long long* accs) {
  if (threadIdx.x < 64) accs[threadIdx.x] = 0ull;
}

__global__ void binarize_weights(const float* __restrict__ w,
                                 signed char* __restrict__ wb) {
  int idx = blockIdx.x * blockDim.x + threadIdx.x;
  if (idx >= TAPS * COUT * CIN) return;
  int c   = idx & 31;
  int o   = (idx >> 5) & 31;
  int tap = idx >> 10;
  signed char v = 0;
  if (tap < 9) {
    float f = w[(o * CIN + c) * 9 + tap];
    v = (f > 0.f) ? 1 : ((f < 0.f) ? -1 : 0);
  }
  wb[idx] = v;
}

__device__ __forceinline__ signed char bsign(float f) {
  return (f > 0.f) ? 1 : ((f < 0.f) ? -1 : 0);
}

__global__ __launch_bounds__(THREADS)
void bconv_wmma(const float* __restrict__ x, const signed char* __restrict__ wb,
                float* __restrict__ out, unsigned long long* __restrict__ accs) {
  __shared__ signed char xb[3 * XCOLS * CIN];
  __shared__ signed char wbs[TAPS * COUT * CIN];
  __shared__ int ytile[COUT * HW];
  __shared__ int psum[WAVES * COUT];
  __shared__ int psq[WAVES * COUT];

  const int h   = blockIdx.x;
  const int n   = blockIdx.y;
  const int tid = threadIdx.x;

#ifdef HAVE_ASYNC_LDS
  {
    for (int i = tid; i < (TAPS * COUT * CIN) / 16; i += THREADS) {
      __builtin_amdgcn_global_load_async_to_lds_b128(
          (vint4_t __attribute__((address_space(1)))*)(wb + i * 16),
          (vint4_t __attribute__((address_space(3)))*)(wbs + i * 16),
          0  , 0  );
    }
    asm volatile("s_wait_asynccnt 0x0" ::: "memory");
  }
#else
  for (int i = tid; i < (TAPS * COUT * CIN) / 16; i += THREADS)
    ((int4*)wbs)[i] = ((const int4*)wb)[i];
#endif

  const float4* x4 = (const float4*)x;
  for (int i = tid; i < 3 * CIN * (HW / 4); i += THREADS) {
    int j  = i % (HW / 4);
    int rc = i / (HW / 4);
    int c  = rc & 31;
    int kh = rc >> 5;
    int hr = h + kh - 1;
    float4 f = {0.f, 0.f, 0.f, 0.f};
    if (hr >= 0 && hr < HW)
      f = x4[((n * CIN + c) * HW + hr) * (HW / 4) + j];
    int colb = (kh * XCOLS + 4 * j + 1) * CIN + c;
    xb[colb]            = bsign(f.x);
    xb[colb + CIN]      = bsign(f.y);
    xb[colb + 2 * CIN]  = bsign(f.z);
    xb[colb + 3 * CIN]  = bsign(f.w);
  }
  if (tid < 3 * CIN * 2) {
    int c    = tid & 31;
    int rc   = tid >> 5;
    int kh   = rc >> 1;
    int col  = (rc & 1) ? (XCOLS - 1) : 0;
    xb[(kh * XCOLS + col) * CIN + c] = 0;
  }
  __syncthreads();

  const int lane = tid & 31;
  const int wv   = tid >> 5;
  const int m    = lane & 15;
  const int hi   = lane >> 4;

  v8i acc0 = {0, 0, 0, 0, 0, 0, 0, 0};
  v8i acc1 = {0, 0, 0, 0, 0, 0, 0, 0};

#pragma unroll
  for (int chunk = 0; chunk < 5; ++chunk) {
    v8i a;
#pragma unroll
    for (int v = 0; v < 8; ++v) {
      int kb  = ((v & 1) << 2) + (((v >> 1) & 1) << 4) + ((v >> 2) << 5) + (hi << 3);
      int k   = chunk * 64 + kb;
      int tap = k >> 5;
      unsigned dw = 0;
      if (tap < 9) {
        int kw  = tap % 3;
        int kh  = tap / 3;
        int col = wv * 16 + m + kw;
        dw = *(const unsigned*)&xb[(kh * XCOLS + col) * CIN + (k & 31)];
      }
      a[v] = (int)dw;
    }
    v8i b0, b1;
#pragma unroll
    for (int v = 0; v < 8; ++v) {
      int K   = ((v & 1) << 2) + (((v >> 1) & 1) << 4) + ((v >> 2) << 5) + (hi << 3);
      int k   = chunk * 64 + K;
      int tap = k >> 5;
      int c0  = k & 31;
      b0[v] = *(const int*)&wbs[(tap * COUT + (lane & 15)) * CIN + c0];
      b1[v] = *(const int*)&wbs[(tap * COUT + 16 + (lane & 15)) * CIN + c0];
    }
    acc0 = __builtin_amdgcn_wmma_i32_16x16x64_iu8(true, a, true, b0, acc0, false, false);
    acc1 = __builtin_amdgcn_wmma_i32_16x16x64_iu8(true, a, true, b1, acc1, false, false);
  }

#pragma unroll
  for (int r = 0; r < 8; ++r) {
    int px = wv * 16 + r + (hi << 3);
    ytile[(lane & 15) * HW + px]        = acc0[r];
    ytile[((lane & 15) + 16) * HW + px] = acc1[r];
  }
  __syncthreads();

  {
    float4* out4 = (float4*)out;
    for (int i = tid; i < COUT * (HW / 4); i += THREADS) {
      int o = i / (HW / 4);
      int j = i - o * (HW / 4);
      const int* yt = &ytile[o * HW + 4 * j];
      float4 v = {(float)yt[0], (float)yt[1], (float)yt[2], (float)yt[3]};
      out4[((n * COUT + o) * HW + h) * (HW / 4) + j] = v;
    }
  }

  {
    int o = tid & 31;
    int seg = tid >> 5;
    int s = 0, q = 0;
#pragma unroll
    for (int i = 0; i < 16; ++i) {
      int v = ytile[o * HW + seg * 16 + i];
      s += v;
      q += v * v;
    }
    psum[seg * 32 + o] = s;
    psq[seg * 32 + o]  = q;
  }
  __syncthreads();
  if (tid < 32) {
    long long S = 0, Q = 0;
#pragma unroll
    for (int seg = 0; seg < WAVES; ++seg) {
      S += psum[seg * 32 + tid];
      Q += psq[seg * 32 + tid];
    }
    atomicAdd(&accs[tid],      (unsigned long long)S);
    atomicAdd(&accs[32 + tid], (unsigned long long)Q);
  }
}

__global__ void bn_finalize(const unsigned long long* __restrict__ accs,
                            const float* __restrict__ gamma,
                            const float* __restrict__ beta,
                            float* __restrict__ sc) {
  int o = threadIdx.x;
  if (o >= 32) return;
  double cnt  = (double)NB * HW * HW;
  double S    = (double)(long long)accs[o];
  double Q    = (double)(long long)accs[32 + o];
  double mean = S / cnt;
  double var  = Q / cnt - mean * mean;
  float inv   = rsqrtf((float)var + BN_EPS);
  float scale = gamma[o] * inv;
  sc[o]      = scale;
  sc[32 + o] = beta[o] - (float)mean * scale;
}

__global__ void bn_apply(float* __restrict__ out, const float* __restrict__ sc,
                         int total4) {
  int i = blockIdx.x * blockDim.x + threadIdx.x;
  if (i >= total4) return;
  float4 v = ((float4*)out)[i];
  int c = (i / (HW * HW / 4)) & 31;
  float s = sc[c], b = sc[32 + c];
  v.x = v.x * s + b;
  v.y = v.y * s + b;
  v.z = v.z * s + b;
  v.w = v.w * s + b;
  ((float4*)out)[i] = v;
}

extern "C" void kernel_launch(void* const* d_in, const int* in_sizes, int n_in,
                              void* d_out, int out_size, void* d_ws, size_t ws_size,
                              hipStream_t stream) {
  const float* x     = (const float*)d_in[0];
  const float* w     = (const float*)d_in[1];
  const float* gamma = (const float*)d_in[2];
  const float* beta  = (const float*)d_in[3];
  float* out = (float*)d_out;

  unsigned long long* accs = (unsigned long long*)d_ws;
  float* sc       = (float*)((char*)d_ws + 512);
  signed char* wb = (signed char*)d_ws + 1024;

  init_accums<<<1, 64, 0, stream>>>(accs);
  binarize_weights<<<(TAPS * COUT * CIN + 255) / 256, 256, 0, stream>>>(w, wb);

  dim3 grid(HW, NB);
  bconv_wmma<<<grid, THREADS, 0, stream>>>(x, wb, out, accs);

  bn_finalize<<<1, 32, 0, stream>>>(accs, gamma, beta, sc);

  int total4 = out_size / 4;
  bn_apply<<<(total4 + 255) / 256, 256, 0, stream>>>(out, sc, total4);
}
